// MultiHeadAttention_59158879535186
// MI455X (gfx1250) — matrix operand order repaired and checked by reading, not compiled, not run on hardware
//
#include <hip/hip_runtime.h>
#include <hip/hip_bf16.h>
#include <cstdint>

// ---------------------------------------------------------------------------
// MHA forward for MI455X (gfx1250, wave32, WMMA, TDM/async staging).
//   k_qkv  : x@W{q,k,v}+b, RoPE on q/k, 1/sqrt(d) folded into q.
//            Q,K stored f16 [B,H,S,D]; V stored TRANSPOSED f16 [B,H,D,S].
//   k_attn : flash attention; K/V^T tiles staged into LDS by the Tensor Data
//            Mover (descriptor-driven DMA, TENSORcnt) when available.
//   k_oproj: attn_out @ Wo + bo -> fp32; A-tile staged via async LDS loads.
// ---------------------------------------------------------------------------

typedef __attribute__((ext_vector_type(16))) _Float16 v16h;
typedef __attribute__((ext_vector_type(8)))  _Float16 v8h;
typedef __attribute__((ext_vector_type(8)))  float    v8f;
typedef __attribute__((ext_vector_type(4)))  unsigned int u32x4;
typedef __attribute__((ext_vector_type(8)))  int      i32x8;
typedef __attribute__((ext_vector_type(4)))  int      i32x4;
typedef int v4i_vs __attribute__((vector_size(4 * sizeof(int))));

constexpr int Bb = 2;
constexpr int Ss = 2048;
constexpr int Ee = 1024;
constexpr int Hh = 16;
constexpr int Dd = 64;
constexpr int Mm = Bb * Ss;
constexpr float LOG2_10000_OVER_32 = 0.415241011860830990f; // log2(10000)/32

#if __has_builtin(__builtin_amdgcn_tensor_load_to_lds)
#define HAVE_TDM 1
#else
#define HAVE_TDM 0
#endif
#if __has_builtin(__builtin_amdgcn_global_load_async_to_lds_b128)
#define HAVE_ASYNC 1
#else
#define HAVE_ASYNC 0
#endif

// Build a 16x32 f16 A/B fragment from two contiguous 16-byte chunks.
// A layout (ISA 7.12.2): lanes 0-15 hold K={0..7,16..23}, lanes 16-31 K={8..15,24..31}.
__device__ inline v16h ld_frag16(const _Float16* p0, const _Float16* p1) {
  v8h lo = *(const v8h*)p0;
  v8h hi = *(const v8h*)p1;
  v16h r;
#pragma unroll
  for (int i = 0; i < 8; ++i) { r[i] = lo[i]; r[i + 8] = hi[i]; }
  return r;
}

__device__ inline v8f wmma_f16(v16h a, v16h b, v8f c) {
  return __builtin_amdgcn_wmma_f32_16x16x32_f16(false, a, false, b,
                                                (short)0, c, false, false);
}

#if HAVE_TDM
// TDM: load a 2-D tile (tile_w x tile_h, 2-byte elems, tile_w==64) into LDS
// with hardware row padding of 4 DWORDs per 32 DWORDs -> LDS row stride 72
// halves (bank-conflict-free, matches the ds_load_b128 fragment reads).
// D# fields per CDNA5 ISA.
__device__ inline void tdm_load_2d(uint32_t lds_byte, uint64_t gaddr,
                                   uint32_t stride_elems, uint32_t tile_w,
                                   uint32_t tile_h) {
  u32x4 g0;
  g0[0] = 1u;                                        // count=1 (user descriptor)
  g0[1] = lds_byte;                                  // lds_addr
  g0[2] = (uint32_t)(gaddr & 0xFFFFFFFFu);           // global_addr[31:0]
  g0[3] = (uint32_t)((gaddr >> 32) & 0x01FFFFFFu)    // global_addr[56:32]
        | (2u << 30);                                // type=2 ("image")
  i32x8 g1;
  g1[0] = (int)((1u << 16)      // data_size = 2 bytes
              | (1u << 20)      // pad_enable
              | (4u << 22)      // pad_interval: 32 DWORDs (one 64-half row)
              | (3u << 25));    // pad_amount: 4 DWORDs (8 halves)
  g1[1] = (int)(tile_w << 16);  // tensor_dim0[15:0] in bits 63:48
  g1[2] = (int)(tile_h << 16);  // tensor_dim1[15:0] in bits 95:80
  g1[3] = (int)(tile_w << 16);  // tile_dim0 in bits 127:112
  g1[4] = (int)tile_h;          // tile_dim1 in bits 143:128
  g1[5] = (int)stride_elems;    // tensor_dim0_stride[31:0]
  g1[6] = 0;
  g1[7] = 0;
  i32x4 z4 = {0, 0, 0, 0};
#if defined(__clang_major__) && __clang_major__ >= 23
  i32x8 z8 = {0, 0, 0, 0, 0, 0, 0, 0};
  __builtin_amdgcn_tensor_load_to_lds(g0, g1, z4, z4, z8, 0);
#else
  __builtin_amdgcn_tensor_load_to_lds(g0, g1, z4, z4, 0);
#endif
}
#endif

#if HAVE_ASYNC
__device__ inline void async_copy16(const void* g, void* l) {
  __builtin_amdgcn_global_load_async_to_lds_b128(
      (__attribute__((address_space(1))) v4i_vs*)(uintptr_t)g,
      (__attribute__((address_space(3))) v4i_vs*)(uint32_t)(uintptr_t)l,
      0, 0);
}
__device__ inline void async_wait0() {
#if __has_builtin(__builtin_amdgcn_s_wait_asynccnt)
  __builtin_amdgcn_s_wait_asynccnt(0);
#else
  asm volatile("s_wait_asynccnt 0" ::: "memory");
#endif
}
#endif

// ---------------------------------------------------------------------------
// Kernel 1: QKV projection + bias + RoPE.  grid = (Mm/128, Hh, 3)
// ---------------------------------------------------------------------------
__global__ void __launch_bounds__(256)
k_qkv(const float* __restrict__ x,
      const float* __restrict__ Wq, const float* __restrict__ Wk,
      const float* __restrict__ Wv,
      const float* __restrict__ bq, const float* __restrict__ bk,
      const float* __restrict__ bv,
      _Float16* __restrict__ Qo, _Float16* __restrict__ Ko,
      _Float16* __restrict__ Vo) {
  __shared__ _Float16 sA[128 * 40];
  __shared__ _Float16 sWt[64 * 40];

  const int tid  = threadIdx.x;
  const int wid  = tid >> 5;
  const int lane = tid & 31;
  const int mb   = blockIdx.x * 128;
  const int h    = blockIdx.y;
  const int nb   = h * 64;
  const int z    = blockIdx.z;        // 0=Q 1=K 2=V

  const float* W    = (z == 0) ? Wq : (z == 1) ? Wk : Wv;
  const float* bias = (z == 0) ? bq : (z == 1) ? bk : bv;
  _Float16*    outp = (z == 0) ? Qo : (z == 1) ? Ko : Vo;

  v8f acc[4] = {};

  for (int kb = 0; kb < Ee; kb += 32) {
    __syncthreads();
#pragma unroll
    for (int j = 0; j < 4; ++j) {       // x tile: 128x32 fp32 -> f16
      int i = tid + 256 * j;
      int r = i >> 3, c4 = i & 7;
      float4 v = *(const float4*)(x + (size_t)(mb + r) * Ee + kb + c4 * 4);
      _Float16* d = &sA[r * 40 + c4 * 4];
      d[0] = (_Float16)v.x; d[1] = (_Float16)v.y;
      d[2] = (_Float16)v.z; d[3] = (_Float16)v.w;
    }
#pragma unroll
    for (int j = 0; j < 2; ++j) {       // W tile, transposed: sWt[n][k]
      int i = tid + 256 * j;
      int r = i >> 4, c4 = i & 15;
      float4 v = *(const float4*)(W + (size_t)(kb + r) * Ee + nb + c4 * 4);
      sWt[(c4 * 4 + 0) * 40 + r] = (_Float16)v.x;
      sWt[(c4 * 4 + 1) * 40 + r] = (_Float16)v.y;
      sWt[(c4 * 4 + 2) * 40 + r] = (_Float16)v.z;
      sWt[(c4 * 4 + 3) * 40 + r] = (_Float16)v.w;
    }
    if (kb + 32 < Ee)
      __builtin_prefetch(W + (size_t)(kb + 32 + (tid >> 3)) * Ee + nb + (tid & 7) * 8, 0, 1);
    __syncthreads();

    const int arow = wid * 16 + (lane & 15);
    const int ak   = (lane < 16) ? 0 : 8;
    v16h a = ld_frag16(&sA[arow * 40 + ak], &sA[arow * 40 + ak + 16]);
#pragma unroll
    for (int t = 0; t < 4; ++t) {
      int bn  = t * 16 + (lane & 15);
      int bk0 = (lane < 16) ? 0 : 8;
      v16h b = ld_frag16(&sWt[bn * 40 + bk0], &sWt[bn * 40 + bk0 + 16]);
      acc[t] = wmma_f16(a, b, acc[t]);
    }
  }

  const int colbase = lane & 15;
  const int half    = (lane < 16) ? 0 : 8;
  float y[4][8];
#pragma unroll
  for (int t = 0; t < 4; ++t) {
    float bv_ = bias[nb + t * 16 + colbase];
#pragma unroll
    for (int r = 0; r < 8; ++r) y[t][r] = acc[t][r] + bv_;
  }
  if (z < 2) {  // RoPE: pair (d, d+32) = tiles (t, t+2), same lane.
#pragma unroll
    for (int tp = 0; tp < 2; ++tp) {
      int jj = tp * 16 + colbase;
      float invf = exp2f(-(float)jj * LOG2_10000_OVER_32);
#pragma unroll
      for (int r = 0; r < 8; ++r) {
        int srow  = (mb + wid * 16 + half + r) & (Ss - 1);
        float ang = (float)srow * invf;
        float cs = cosf(ang), sn = sinf(ang);
        float lo = y[tp][r], hi = y[tp + 2][r];
        y[tp][r]     = lo * cs - hi * sn;
        y[tp + 2][r] = hi * cs + lo * sn;
      }
    }
  }
  if (z == 0) {
#pragma unroll
    for (int t = 0; t < 4; ++t)
#pragma unroll
      for (int r = 0; r < 8; ++r) y[t][r] *= 0.125f;
  }
#pragma unroll
  for (int t = 0; t < 4; ++t) {
#pragma unroll
    for (int r = 0; r < 8; ++r) {
      int row = mb + wid * 16 + half + r;
      int bbv = row >> 11, s = row & (Ss - 1);
      int d = t * 16 + colbase;
      if (z == 2)  // V stored transposed: [B,H,D,S] so attention can DMA it
        outp[(size_t)((bbv * Hh + h) * Dd + d) * Ss + s] = (_Float16)y[t][r];
      else         // Q,K: [B,H,S,D]
        outp[(size_t)((bbv * Hh + h) * Ss + s) * Dd + d] = (_Float16)y[t][r];
    }
  }
}

// ---------------------------------------------------------------------------
// Kernel 2: flash attention.  grid = (Ss/128, Hh, Bb), 8 waves x 16 q-rows.
// K and V^T tiles are staged by the Tensor Data Mover when available.
// ---------------------------------------------------------------------------
__global__ void __launch_bounds__(256)
k_attn(const _Float16* __restrict__ Q, const _Float16* __restrict__ Kp,
       const _Float16* __restrict__ Vt, const int* __restrict__ amask,
       _Float16* __restrict__ O) {
  __shared__ _Float16 sK[64 * 72];       // K tile, row-major [key][d], stride 72
  __shared__ _Float16 sVt[64 * 72];      // V^T tile [d][key], stride 72
  __shared__ _Float16 sP[8 * 16 * 72];   // per-wave P scratch

  const int tid = threadIdx.x, wid = tid >> 5, lane = tid & 31;
  const int bbv = blockIdx.z, h = blockIdx.y, qb = blockIdx.x * 128;
  const _Float16* Qh  = Q  + (size_t)((bbv * Hh + h) * Ss) * Dd;
  const _Float16* Kh  = Kp + (size_t)((bbv * Hh + h) * Ss) * Dd;
  const _Float16* Vth = Vt + (size_t)((bbv * Hh + h) * Dd) * Ss;

  const int qrow = qb + wid * 16 + (lane & 15);
  v16h qf[2];
#pragma unroll
  for (int c = 0; c < 2; ++c) {
    int k0 = c * 32 + ((lane < 16) ? 0 : 8);
    qf[c] = ld_frag16(Qh + (size_t)qrow * Dd + k0,
                      Qh + (size_t)qrow * Dd + k0 + 16);
  }

  float mold[8], lsum[8];
#pragma unroll
  for (int r = 0; r < 8; ++r) { mold[r] = -1e30f; lsum[r] = 0.0f; }
  v8f oacc[4] = {};

  const int myrowbase = qb + wid * 16 + ((lane < 16) ? 0 : 8);
  const int nkt = (qb >> 6) + 2;

  for (int kt = 0; kt < nkt; ++kt) {
    const int kb = kt * 64;
    __syncthreads();   // all LDS reads of previous tile complete
#if HAVE_TDM
    if (tid < 32) {    // wave 0 drives the Tensor Data Mover
      tdm_load_2d((uint32_t)(uintptr_t)&sK[0],
                  (uint64_t)(uintptr_t)(Kh + (size_t)kb * Dd),
                  (uint32_t)Dd, 64u, 64u);
      tdm_load_2d((uint32_t)(uintptr_t)&sVt[0],
                  (uint64_t)(uintptr_t)(Vth + kb),
                  (uint32_t)Ss, 64u, 64u);
      __builtin_amdgcn_s_wait_tensorcnt(0);
    }
#else
#pragma unroll
    for (int j = 0; j < 2; ++j) {       // K tile
      int i = tid + 256 * j;
      int r = i >> 3, c8 = i & 7;
      *(v8h*)&sK[r * 72 + c8 * 8] =
          *(const v8h*)(Kh + (size_t)(kb + r) * Dd + c8 * 8);
    }
#pragma unroll
    for (int j = 0; j < 2; ++j) {       // V^T tile (already transposed)
      int i = tid + 256 * j;
      int r = i >> 3, c8 = i & 7;
      *(v8h*)&sVt[r * 72 + c8 * 8] =
          *(const v8h*)(Vth + (size_t)r * Ss + kb + c8 * 8);
    }
#endif
    __syncthreads();

    // S = Q K^T
    v8f sacc[4] = {};
#pragma unroll
    for (int c = 0; c < 2; ++c) {
#pragma unroll
      for (int t = 0; t < 4; ++t) {
        int krow = t * 16 + (lane & 15);
        int k0   = c * 32 + ((lane < 16) ? 0 : 8);
        v16h bf = ld_frag16(&sK[krow * 72 + k0], &sK[krow * 72 + k0 + 16]);
        sacc[t] = wmma_f16(qf[c], bf, sacc[t]);
      }
    }

    // Causal + padding mask, online softmax.
    int col[4], keep[4];
#pragma unroll
    for (int t = 0; t < 4; ++t) {
      col[t]  = kb + t * 16 + (lane & 15);
      keep[t] = amask[bbv * Ss + col[t]];
    }
    float mnew[8], fsc[8];
#pragma unroll
    for (int r = 0; r < 8; ++r) {
      int row = myrowbase + r;
      float mx = -1e30f;
#pragma unroll
      for (int t = 0; t < 4; ++t) {
        float sv = sacc[t][r];
        if (col[t] > row || keep[t] == 0) { sv = -1e30f; sacc[t][r] = sv; }
        mx = fmaxf(mx, sv);
      }
#pragma unroll
      for (int dl = 1; dl < 16; dl <<= 1)
        mx = fmaxf(mx, __shfl_xor(mx, dl, 32));
      mnew[r] = fmaxf(mold[r], mx);
      fsc[r]  = __expf(mold[r] - mnew[r]);
      mold[r] = mnew[r];
    }
#pragma unroll
    for (int r = 0; r < 8; ++r) {
      float sum = 0.0f;
#pragma unroll
      for (int t = 0; t < 4; ++t) {
        float p = __expf(sacc[t][r] - mnew[r]);
        sacc[t][r] = p;
        sum += p;
      }
#pragma unroll
      for (int dl = 1; dl < 16; dl <<= 1) sum += __shfl_xor(sum, dl, 32);
      lsum[r] = lsum[r] * fsc[r] + sum;
    }
#pragma unroll
    for (int t = 0; t < 4; ++t)
#pragma unroll
      for (int r = 0; r < 8; ++r) oacc[t][r] *= fsc[r];

    // C-layout P -> A-layout via per-wave LDS scratch.
    _Float16* pw = &sP[wid * 16 * 72];
    const int prow = (lane < 16) ? 0 : 8;
#pragma unroll
    for (int t = 0; t < 4; ++t)
#pragma unroll
      for (int r = 0; r < 8; ++r)
        pw[(prow + r) * 72 + t * 16 + (lane & 15)] = (_Float16)sacc[t][r];

    // O += P V
#pragma unroll
    for (int c = 0; c < 2; ++c) {
      int k0 = c * 32 + ((lane < 16) ? 0 : 8);
      const _Float16* pr = &pw[(lane & 15) * 72];
      v16h pa = ld_frag16(pr + k0, pr + k0 + 16);
#pragma unroll
      for (int t = 0; t < 4; ++t) {
        int vn  = t * 16 + (lane & 15);
        int vk0 = c * 32 + ((lane < 16) ? 0 : 8);
        v16h vb = ld_frag16(&sVt[vn * 72 + vk0], &sVt[vn * 72 + vk0 + 16]);
        oacc[t] = wmma_f16(pa, vb, oacc[t]);
      }
    }
  }

#pragma unroll
  for (int r = 0; r < 8; ++r) {
    float inv = 1.0f / lsum[r];
    int row = myrowbase + r;
#pragma unroll
    for (int t = 0; t < 4; ++t) {
      O[(size_t)(bbv * Ss + row) * Ee + h * 64 + t * 16 + (lane & 15)] =
          (_Float16)(oacc[t][r] * inv);
    }
  }
}

// ---------------------------------------------------------------------------
// Kernel 3: output projection.  grid = (Mm/128, Ee/64)
// ---------------------------------------------------------------------------
__global__ void __launch_bounds__(256)
k_oproj(const _Float16* __restrict__ Oin, const float* __restrict__ Wo,
        const float* __restrict__ bo, float* __restrict__ out) {
  __shared__ _Float16 sA[128 * 40];
  __shared__ _Float16 sWt[64 * 40];

  const int tid = threadIdx.x, wid = tid >> 5, lane = tid & 31;
  const int mb = blockIdx.x * 128, nb = blockIdx.y * 64;

  v8f acc[4] = {};
  for (int kb = 0; kb < Ee; kb += 32) {
    __syncthreads();
#if HAVE_ASYNC
#pragma unroll
    for (int j = 0; j < 2; ++j) {       // A tile via async LDS loads
      int i = tid + 256 * j;
      int r = i >> 2, c8 = i & 3;
      async_copy16(Oin + (size_t)(mb + r) * Ee + kb + c8 * 8,
                   &sA[r * 40 + c8 * 8]);
    }
#else
#pragma unroll
    for (int j = 0; j < 2; ++j) {
      int i = tid + 256 * j;
      int r = i >> 2, c8 = i & 3;
      *(v8h*)&sA[r * 40 + c8 * 8] =
          *(const v8h*)(Oin + (size_t)(mb + r) * Ee + kb + c8 * 8);
    }
#endif
#pragma unroll
    for (int j = 0; j < 2; ++j) {       // Wo tile, transposed + f16 convert
      int i = tid + 256 * j;
      int r = i >> 4, c4 = i & 15;
      float4 v = *(const float4*)(Wo + (size_t)(kb + r) * Ee + nb + c4 * 4);
      sWt[(c4 * 4 + 0) * 40 + r] = (_Float16)v.x;
      sWt[(c4 * 4 + 1) * 40 + r] = (_Float16)v.y;
      sWt[(c4 * 4 + 2) * 40 + r] = (_Float16)v.z;
      sWt[(c4 * 4 + 3) * 40 + r] = (_Float16)v.w;
    }
    if (kb + 32 < Ee)
      __builtin_prefetch(Wo + (size_t)(kb + 32 + (tid >> 3)) * Ee + nb + (tid & 7) * 8, 0, 1);
#if HAVE_ASYNC
    async_wait0();
#endif
    __syncthreads();

    const int arow = wid * 16 + (lane & 15);
    const int ak   = (lane < 16) ? 0 : 8;
    v16h a = ld_frag16(&sA[arow * 40 + ak], &sA[arow * 40 + ak + 16]);
#pragma unroll
    for (int t = 0; t < 4; ++t) {
      int bn  = t * 16 + (lane & 15);
      int bk0 = (lane < 16) ? 0 : 8;
      v16h b = ld_frag16(&sWt[bn * 40 + bk0], &sWt[bn * 40 + bk0 + 16]);
      acc[t] = wmma_f16(a, b, acc[t]);
    }
  }

  const int colbase = lane & 15;
  const int half    = (lane < 16) ? 0 : 8;
#pragma unroll
  for (int t = 0; t < 4; ++t) {
    float bv_ = bo[nb + t * 16 + colbase];
#pragma unroll
    for (int r = 0; r < 8; ++r) {
      int row = mb + wid * 16 + half + r;
      out[(size_t)row * Ee + nb + t * 16 + colbase] = acc[t][r] + bv_;
    }
  }
}

// ---------------------------------------------------------------------------
extern "C" void kernel_launch(void* const* d_in, const int* in_sizes, int n_in,
                              void* d_out, int out_size, void* d_ws,
                              size_t ws_size, hipStream_t stream) {
  (void)in_sizes; (void)n_in; (void)out_size; (void)ws_size;
  const float* x  = (const float*)d_in[0];
  const int* am   = (const int*)d_in[1];
  const float* Wq = (const float*)d_in[2];
  const float* bq = (const float*)d_in[3];
  const float* Wk = (const float*)d_in[4];
  const float* bk = (const float*)d_in[5];
  const float* Wv = (const float*)d_in[6];
  const float* bv = (const float*)d_in[7];
  const float* Wo = (const float*)d_in[8];
  const float* bo = (const float*)d_in[9];
  float* out = (float*)d_out;

  _Float16* ws = (_Float16*)d_ws;
  const size_t per = (size_t)Bb * Hh * Ss * Dd;
  _Float16* Qw  = ws;
  _Float16* Kw  = ws + per;
  _Float16* Vtw = ws + 2 * per;   // [B,H,D,S]
  _Float16* Ow  = ws + 3 * per;   // [B,S,E]

  dim3 blk(256);
  k_qkv<<<dim3(Mm / 128, Hh, 3), blk, 0, stream>>>(x, Wq, Wk, Wv, bq, bk, bv,
                                                   Qw, Kw, Vtw);
  k_attn<<<dim3(Ss / 128, Hh, Bb), blk, 0, stream>>>(Qw, Kw, Vtw, am, Ow);
  k_oproj<<<dim3(Mm / 128, Ee / 64), blk, 0, stream>>>(Ow, Wo, bo, out);
}
